// CausalSelfAttention_54984171323734
// MI455X (gfx1250) — matrix operand order repaired and checked by reading, not compiled, not run on hardware
//
#include <hip/hip_runtime.h>
#include <hip/hip_bf16.h>

// ---------------------------------------------------------------------------
// CausalSelfAttention  B=2 L=2048 HID=2048 H=16 D=128  (MI455X / gfx1250)
// bf16 WMMA (v_wmma_f32_16x16x32_bf16) for all matmuls, fp32 softmax.
// All operand tiles are staged into LDS with global_load_async_to_lds_b128,
// double-buffered so the async stage of step t+1 overlaps compute of step t.
// ---------------------------------------------------------------------------

typedef __bf16 bf16x16 __attribute__((ext_vector_type(16)));
typedef __bf16 bf16x8  __attribute__((ext_vector_type(8)));
typedef __bf16 bf16x4  __attribute__((ext_vector_type(4)));
typedef float  floatx8 __attribute__((ext_vector_type(8)));

union ABfrag { bf16x16 v; bf16x8 h[2]; };

static constexpr int Bc  = 2;
static constexpr int Lc  = 2048;
static constexpr int HID = 2048;
static constexpr int Hc  = 16;
static constexpr int Dc  = 128;

__device__ __forceinline__ __bf16 f2bf(float f) {
  unsigned u = __float_as_uint(f);
  unsigned r = (u + 0x7FFFu + ((u >> 16) & 1u)) >> 16;   // RNE
  unsigned short s = (unsigned short)r;
  __bf16 b;
  __builtin_memcpy(&b, &s, 2);
  return b;
}
__device__ __forceinline__ float bf2f(__bf16 b) {
  unsigned short s;
  __builtin_memcpy(&s, &b, 2);
  return __uint_as_float(((unsigned)s) << 16);
}

// Async global->LDS copy, 16 bytes per lane (ASYNCcnt-tracked).
__device__ __forceinline__ void async_b128(const __bf16* g, unsigned lds_off) {
  asm volatile("global_load_async_to_lds_b128 %0, %1, off"
               :: "v"(lds_off), "v"(g) : "memory");
}

// ---------------------------------------------------------------------------
// fp32 -> bf16 conversion (vectorized 4-wide)
// ---------------------------------------------------------------------------
__global__ __launch_bounds__(256)
void csa_cvt_f32_bf16(const float* __restrict__ src, __bf16* __restrict__ dst, int n) {
  int i = (blockIdx.x * 256 + threadIdx.x) * 4;
  if (i < n) {
    float4 f = *(const float4*)(src + i);
    bf16x4 o;
    o[0] = f2bf(f.x); o[1] = f2bf(f.y); o[2] = f2bf(f.z); o[3] = f2bf(f.w);
    *(bf16x4*)(dst + i) = o;
  }
}

// ---------------------------------------------------------------------------
// GEMM: C(M,N) = A(M,K) @ W(N,K)^T   — bf16 in, fp32 accum.
// WG = 8 waves (2M x 4N), each wave 64x64 (4x4 WMMA tiles) -> 128x256 per WG.
// A (128x32) and B (256x32) k-slices are async-staged into padded LDS,
// double-buffered; fragments are read back with ds_load_b128.
// MODE 0: store bf16 into head layout (b,h,l,d)            [Q, K]
// MODE 1: store bf16 into transposed head layout (b,h,d,l) [V]
// MODE 2: store fp32 row-major (M,N)                       [final output]
// ---------------------------------------------------------------------------
static constexpr int G_LDA = 40;     // 32 + 8 pad -> 20-dword stride, no conflicts

template<int MODE>
__global__ __launch_bounds__(256)
void csa_gemm_bf16(const __bf16* __restrict__ A, const __bf16* __restrict__ W,
                   void* __restrict__ Cout, int M, int N, int K) {
  __shared__ __bf16 As[2][128 * G_LDA];   // 2 x 10 KB
  __shared__ __bf16 Bs[2][256 * G_LDA];   // 2 x 20 KB

  const int tid  = threadIdx.x;
  const int lane = tid & 31;
  const int wave = tid >> 5;
  const int wm = wave & 1, wn = wave >> 1;
  const int lh = lane >> 4;          // lane half (0/1)
  const int lm = lane & 15;
  const int mBlk = blockIdx.x * 128;
  const int nBlk = blockIdx.y * 256;

  const unsigned aOff = (unsigned)(uintptr_t)(&As[0][0]);
  const unsigned bOff = (unsigned)(uintptr_t)(&Bs[0][0]);

  auto stage = [&](int kb, int sel) {
    const unsigned ao = aOff + (unsigned)sel * (128 * G_LDA * 2);
    const unsigned bo = bOff + (unsigned)sel * (256 * G_LDA * 2);
#pragma unroll
    for (int pass = 0; pass < 2; ++pass) {     // A: 512 x 16B chunks
      const int c = tid + pass * 256;
      const int r = c >> 2, o = (c & 3) * 8;
      async_b128(A + (size_t)(mBlk + r) * K + kb + o,
                 ao + (unsigned)(r * G_LDA + o) * 2u);
    }
#pragma unroll
    for (int pass = 0; pass < 4; ++pass) {     // B: 1024 x 16B chunks
      const int c = tid + pass * 256;
      const int r = c >> 2, o = (c & 3) * 8;
      async_b128(W + (size_t)(nBlk + r) * K + kb + o,
                 bo + (unsigned)(r * G_LDA + o) * 2u);
    }
  };

  floatx8 acc[4][4] = {};

  stage(0, 0);
  const int T = K >> 5;
  for (int t = 0; t < T; ++t) {
    const int cur = t & 1;
    if (t + 1 < T) {
      stage((t + 1) * 32, cur ^ 1);
      asm volatile("s_wait_asynccnt 0x6" ::: "memory");   // slice t landed (6 ops/thread)
    } else {
      asm volatile("s_wait_asynccnt 0x0" ::: "memory");
    }
    __syncthreads();

    const __bf16* Asl = &As[cur][0];
    const __bf16* Bsl = &Bs[cur][0];
    ABfrag bfr[4];
#pragma unroll
    for (int j = 0; j < 4; ++j) {
      const __bf16* bp = Bsl + (wn * 64 + 16 * j + lm) * G_LDA + lh * 16;
      bfr[j].h[0] = *(const bf16x8*)bp;
      bfr[j].h[1] = *(const bf16x8*)(bp + 8);
    }
#pragma unroll
    for (int i = 0; i < 4; ++i) {
      ABfrag af;
      const __bf16* ap = Asl + (wm * 64 + 16 * i + lm) * G_LDA + lh * 16;
      af.h[0] = *(const bf16x8*)ap;
      af.h[1] = *(const bf16x8*)(ap + 8);
#pragma unroll
      for (int j = 0; j < 4; ++j)
        acc[i][j] = __builtin_amdgcn_wmma_f32_16x16x32_bf16(
            false, af.v, false, bfr[j].v, (short)0, acc[i][j], false, false);
    }
    __syncthreads();   // protect buffer `cur` before it is restaged at t+2
  }

  const int m0 = mBlk + wm * 64;
  const int n0 = nBlk + wn * 64;
#pragma unroll
  for (int i = 0; i < 4; ++i)
#pragma unroll
    for (int j = 0; j < 4; ++j)
#pragma unroll
      for (int r = 0; r < 8; ++r) {
        const int m = m0 + 16 * i + r + lh * 8;
        const int n = n0 + 16 * j + lm;
        const float v = acc[i][j][r];
        if (MODE == 0) {
          const int bb = m >> 11, l = m & (Lc - 1);
          const int hh = n >> 7,  d = n & (Dc - 1);
          ((__bf16*)Cout)[(((size_t)bb * Hc + hh) * Lc + l) * Dc + d] = f2bf(v);
        } else if (MODE == 1) {
          const int bb = m >> 11, l = m & (Lc - 1);
          const int hh = n >> 7,  d = n & (Dc - 1);
          ((__bf16*)Cout)[(((size_t)bb * Hc + hh) * Dc + d) * Lc + l] = f2bf(v);
        } else {
          ((float*)Cout)[(size_t)m * N + n] = v;
        }
      }
}

// ---------------------------------------------------------------------------
// In-place RoPE on Q and K, head layout (b,h,l,d). One thread per (d, d+64).
// ---------------------------------------------------------------------------
__global__ __launch_bounds__(256)
void csa_rope(__bf16* __restrict__ Qh, __bf16* __restrict__ Kh,
              const float* __restrict__ cosT, const float* __restrict__ sinT) {
  const int tid = blockIdx.x * 256 + threadIdx.x;
  const int t   = tid >> 22;               // 0: Q, 1: K   (2^22 pairs per tensor)
  const int r   = tid & 4194303;
  const int d   = r & 63;
  const int l   = (r >> 6) & (Lc - 1);
  const int bh  = r >> 17;                 // 0..31
  __bf16* p = (t ? Kh : Qh) + ((size_t)bh * Lc + l) * Dc;
  const float x1 = bf2f(p[d]);
  const float x2 = bf2f(p[d + 64]);
  const float c1 = cosT[l * Dc + d],      s1 = sinT[l * Dc + d];
  const float c2 = cosT[l * Dc + d + 64], s2 = sinT[l * Dc + d + 64];
  p[d]      = f2bf(x1 * c1 - x2 * s1);
  p[d + 64] = f2bf(x2 * c2 + x1 * s2);
}

// ---------------------------------------------------------------------------
// Flash attention. Grid: (B*H, L/128). 8 waves/WG, each wave owns 16 q-rows.
// K (32kv x 128d) and V^T (128d x 32kv) tiles are async-staged into LDS,
// double-buffered, and shared by all 8 waves. S^T = K @ Q^T so the score
// accumulator has q-row == lane; exp(S) re-packs into the P A-fragment
// entirely lane-locally. Softmax bookkeeping runs in the exp2 domain
// (scale and log2(e) folded into one constant -> one mul + native v_exp
// per score), and causal masking is applied only in the <=2 diagonal
// kv-steps per wave (wave-uniform branch).
// ---------------------------------------------------------------------------
static constexpr int KT_LDA = 136;   // 128 + 8 pad -> 68-dword stride, no conflicts
static constexpr int VT_LDA = 40;    //  32 + 8 pad -> 20-dword stride, no conflicts

__global__ __launch_bounds__(256)
void csa_flash(const __bf16* __restrict__ Qh, const __bf16* __restrict__ Kh,
               const __bf16* __restrict__ Vt, __bf16* __restrict__ Obf) {
  __shared__ __bf16 Kt[2][32 * KT_LDA];
  __shared__ __bf16 Vl[2][128 * VT_LDA];

  const int tid  = threadIdx.x;
  const int lane = tid & 31;
  const int wave = tid >> 5;
  const int lh = lane >> 4, lm = lane & 15;
  const int bh = blockIdx.x;
  const int b  = bh >> 4, h = bh & (Hc - 1);
  const int q0 = blockIdx.y * 128 + wave * 16;

  const __bf16* Qb = Qh + (size_t)bh * Lc * Dc;
  const __bf16* Kb = Kh + (size_t)bh * Lc * Dc;
  const __bf16* Vb = Vt + (size_t)bh * Dc * Lc;

  const unsigned ktOff = (unsigned)(uintptr_t)(&Kt[0][0]);
  const unsigned vtOff = (unsigned)(uintptr_t)(&Vl[0][0]);

  auto stage_kv = [&](int kv0, int sel) {
    const unsigned ko = ktOff + (unsigned)sel * (32 * KT_LDA * 2);
    const unsigned vo = vtOff + (unsigned)sel * (128 * VT_LDA * 2);
#pragma unroll
    for (int pass = 0; pass < 2; ++pass) {
      const int c = tid + pass * 256;                 // 512 x 16B chunks each
      const int kr = c >> 4, kc = (c & 15) * 8;
      async_b128(Kb + (size_t)(kv0 + kr) * Dc + kc,
                 ko + (unsigned)(kr * KT_LDA + kc) * 2u);
      const int vd = c >> 2, vc = (c & 3) * 8;
      async_b128(Vb + (size_t)vd * Lc + kv0 + vc,
                 vo + (unsigned)(vd * VT_LDA + vc) * 2u);
    }
  };

  // Q as B-operand fragments (32 d per step, 4 steps covering D=128)
  ABfrag qf[4];
#pragma unroll
  for (int db = 0; db < 4; ++db) {
    const __bf16* qp = Qb + (size_t)(q0 + lm) * Dc + db * 32 + lh * 16;
    qf[db].h[0] = *(const bf16x8*)qp;
    qf[db].h[1] = *(const bf16x8*)(qp + 8);
  }

  floatx8 accO[8] = {};                 // 16 q-rows x 128 d
  // Online softmax state in the exp2 domain: z = s_raw * k2, p = 2^(z - m).
  const float k2 = 0.12751666805037743f;      // log2(e) / sqrt(128)
  float mrow = -1.0e38f, lrow = 0.0f;
  const int qrow = q0 + lm;

  const int T = (blockIdx.y + 1) * 4;   // (blockIdx.y+1)*128 / 32 kv-steps

  stage_kv(0, 0);
  for (int t = 0; t < T; ++t) {
    const int kv0 = t * 32;
    const int cur = t & 1;
    if (t + 1 < T) {
      stage_kv((t + 1) * 32, cur ^ 1);
      asm volatile("s_wait_asynccnt 0x4" ::: "memory");   // tile t landed (4 ops/thread)
    } else {
      asm volatile("s_wait_asynccnt 0x0" ::: "memory");
    }
    __syncthreads();

    if (kv0 < q0 + 16) {
      const __bf16* Ks = &Kt[cur][0];
      const __bf16* Vs = &Vl[cur][0];

      // ---- S^T tiles: two 16(kv) x 16(q) accumulators ------------------
      floatx8 s[2] = {};
#pragma unroll
      for (int t2 = 0; t2 < 2; ++t2)
#pragma unroll
        for (int db = 0; db < 4; ++db) {
          ABfrag kf;
          const __bf16* kp = Ks + (16 * t2 + lm) * KT_LDA + db * 32 + lh * 16;
          kf.h[0] = *(const bf16x8*)kp;
          kf.h[1] = *(const bf16x8*)(kp + 8);
          s[t2] = __builtin_amdgcn_wmma_f32_16x16x32_bf16(
              false, kf.v, false, qf[db].v, (short)0, s[t2], false, false);
        }

      // ---- scores -> exp2 domain; mask only on diagonal blocks ---------
      float z[2][8];
#pragma unroll
      for (int t2 = 0; t2 < 2; ++t2)
#pragma unroll
        for (int r = 0; r < 8; ++r)
          z[t2][r] = s[t2][r] * k2;

      if (kv0 + 31 > q0) {              // wave-uniform: <=2 iterations per wave
#pragma unroll
        for (int t2 = 0; t2 < 2; ++t2)
#pragma unroll
          for (int r = 0; r < 8; ++r) {
            const int kv = kv0 + 16 * t2 + r + lh * 8;
            z[t2][r] = (kv > qrow) ? -1.0e38f : z[t2][r];
          }
      }

      float locmax = -1.0e38f;
#pragma unroll
      for (int t2 = 0; t2 < 2; ++t2)
#pragma unroll
        for (int r = 0; r < 8; ++r) locmax = fmaxf(locmax, z[t2][r]);
      locmax = fmaxf(locmax, __shfl_xor(locmax, 16, 32));
      const float mnew = fmaxf(mrow, locmax);
      const float corr = exp2f(mrow - mnew);
      mrow = mnew;

      // ---- exp2 + lane-local repack into P A-fragment ------------------
      ABfrag pa;
      float rsum = 0.0f;
#pragma unroll
      for (int t2 = 0; t2 < 2; ++t2)
#pragma unroll
        for (int r = 0; r < 8; ++r) {
          const float e = exp2f(z[t2][r] - mnew);
          rsum += e;
          pa.h[t2][r] = f2bf(e);
        }
      rsum += __shfl_xor(rsum, 16, 32);
      lrow = lrow * corr + rsum;

      // ---- rescale O (row scale lives in lane row&15) ------------------
      float cr[8];
#pragma unroll
      for (int r = 0; r < 8; ++r) cr[r] = __shfl(corr, r + lh * 8, 32);
#pragma unroll
      for (int td = 0; td < 8; ++td)
#pragma unroll
        for (int r = 0; r < 8; ++r) accO[td][r] *= cr[r];

      // ---- O += P @ V (V fragments from LDS) ---------------------------
#pragma unroll
      for (int td = 0; td < 8; ++td) {
        ABfrag vf;
        const __bf16* vp = Vs + (16 * td + lm) * VT_LDA + lh * 8;
        vf.h[0] = *(const bf16x8*)vp;
        vf.h[1] = *(const bf16x8*)(vp + 16);
        accO[td] = __builtin_amdgcn_wmma_f32_16x16x32_bf16(
            false, pa.v, false, vf.v, (short)0, accO[td], false, false);
      }
    }
    __syncthreads();   // buffer `cur` is restaged at t+2
  }

  // ---- epilogue: divide by row-sum, store bf16 (b*L+q, h*D+d) ----------
  float inv[8];
#pragma unroll
  for (int r = 0; r < 8; ++r) {
    const float lr = __shfl(lrow, r + lh * 8, 32);
    inv[r] = 1.0f / lr;
  }
#pragma unroll
  for (int td = 0; td < 8; ++td)
#pragma unroll
    for (int r = 0; r < 8; ++r) {
      const int row = q0 + r + lh * 8;
      const int d   = 16 * td + lm;
      Obf[((size_t)b * Lc + row) * HID + h * Dc + d] = f2bf(accO[td][r] * inv[r]);
    }
}

// ---------------------------------------------------------------------------
// Host orchestration
// ---------------------------------------------------------------------------
extern "C" void kernel_launch(void* const* d_in, const int* in_sizes, int n_in,
                              void* d_out, int out_size, void* d_ws, size_t ws_size,
                              hipStream_t stream) {
  const float* X    = (const float*)d_in[0];
  const float* cosT = (const float*)d_in[1];
  const float* sinT = (const float*)d_in[2];
  const float* Wq   = (const float*)d_in[3];
  const float* Wk   = (const float*)d_in[4];
  const float* Wv   = (const float*)d_in[5];
  const float* Wo   = (const float*)d_in[6];

  char* ws = (char*)d_ws;
  __bf16* Xbf = (__bf16*)(ws);                    // 4096x2048 bf16 = 16 MiB
  __bf16* Wqb = (__bf16*)(ws + (16u << 20));      // 8 MiB each
  __bf16* Wkb = (__bf16*)(ws + (24u << 20));
  __bf16* Wvb = (__bf16*)(ws + (32u << 20));
  __bf16* Wob = (__bf16*)(ws + (40u << 20));
  __bf16* Qh  = (__bf16*)(ws + (48u << 20));      // (b,h,l,d) bf16 = 16 MiB
  __bf16* Kh  = (__bf16*)(ws + (64u << 20));
  __bf16* Vt  = (__bf16*)(ws + (80u << 20));      // (b,h,d,l)
  __bf16* Obf = (__bf16*)(ws + (96u << 20));      // (b*l, hid)

  const int nX = Bc * Lc * HID;    // 8388608
  const int nW = HID * HID;        // 4194304
  csa_cvt_f32_bf16<<<nX / 1024, 256, 0, stream>>>(X,  Xbf, nX);
  csa_cvt_f32_bf16<<<nW / 1024, 256, 0, stream>>>(Wq, Wqb, nW);
  csa_cvt_f32_bf16<<<nW / 1024, 256, 0, stream>>>(Wk, Wkb, nW);
  csa_cvt_f32_bf16<<<nW / 1024, 256, 0, stream>>>(Wv, Wvb, nW);
  csa_cvt_f32_bf16<<<nW / 1024, 256, 0, stream>>>(Wo, Wob, nW);

  const dim3 gg(32, 8);    // (M/128, N/256) for M=4096, N=2048
  csa_gemm_bf16<0><<<gg, 256, 0, stream>>>(Xbf, Wqb, Qh, Bc * Lc, HID, HID);
  csa_gemm_bf16<0><<<gg, 256, 0, stream>>>(Xbf, Wkb, Kh, Bc * Lc, HID, HID);
  csa_gemm_bf16<1><<<gg, 256, 0, stream>>>(Xbf, Wvb, Vt, Bc * Lc, HID, HID);

  csa_rope<<<32768, 256, 0, stream>>>(Qh, Kh, cosT, sinT);   // 2 * 4M pairs

  csa_flash<<<dim3(Bc * Hc, Lc / 128), 256, 0, stream>>>(Qh, Kh, Vt, Obf);

  csa_gemm_bf16<2><<<gg, 256, 0, stream>>>(Obf, Wob, d_out, Bc * Lc, HID, HID);
}
